// MolEncoder_30949534335599
// MI455X (gfx1250) — compile-verified, matrix operand order checked by reading, not run on hardware
//
#include <hip/hip_runtime.h>
#include <hip/hip_bf16.h>

#define HID 384
#define LAYERS 4

typedef __attribute__((ext_vector_type(16))) __bf16 v16bf;
typedef __attribute__((ext_vector_type(8)))  __bf16 v8bf;
typedef __attribute__((ext_vector_type(8)))  float  v8f;

__device__ __constant__ int d_atom_off[9] = {0,119,128,139,151,160,165,173,175};
__device__ __constant__ int d_bond_off[3] = {0,22,28};

// ---------------------------------------------------------------------------
// Utility kernels
// ---------------------------------------------------------------------------
__global__ void fill_f32(float* __restrict__ p, float v, size_t n) {
    size_t i = (size_t)blockIdx.x * blockDim.x + threadIdx.x;
    if (i < n) p[i] = v;
}

// W: K x Nn (f32, row-major)  ->  Wt: Nn x K (bf16, row-major)
__global__ void cvt_transpose(const float* __restrict__ W, __bf16* __restrict__ Wt,
                              int K, int Nn) {
    int idx = blockIdx.x * blockDim.x + threadIdx.x;
    if (idx >= K * Nn) return;
    int k = idx / Nn, n = idx % Nn;
    Wt[(size_t)n * K + k] = (__bf16)W[idx];
}

__global__ void add_to_bf16(const float* __restrict__ a, const float* __restrict__ b,
                            __bf16* __restrict__ o, size_t n) {
    size_t i = (size_t)blockIdx.x * blockDim.x + threadIdx.x;
    if (i < n) o[i] = (__bf16)(a[i] + b[i]);
}

// ---------------------------------------------------------------------------
// Embedding sum kernels
// ---------------------------------------------------------------------------
__global__ __launch_bounds__(HID) void embed_atoms(const int* __restrict__ x,
                                                   const float* __restrict__ emb,
                                                   __bf16* __restrict__ out, int N) {
    int n = blockIdx.x, d = threadIdx.x;
    __shared__ int ids[9];
    if (d < 9) ids[d] = x[(size_t)n * 9 + d] + d_atom_off[d];
    __syncthreads();
    float s = 0.f;
    #pragma unroll
    for (int f = 0; f < 9; ++f) s += emb[(size_t)ids[f] * HID + d];
    out[(size_t)n * HID + d] = (__bf16)s;
}

__global__ __launch_bounds__(HID) void embed_bonds(const int* __restrict__ ea,
                                                   const float* __restrict__ emb,
                                                   __bf16* __restrict__ out, int E) {
    int e = blockIdx.x, d = threadIdx.x;
    __shared__ int ids[3];
    if (d < 3) ids[d] = ea[(size_t)e * 3 + d] + d_bond_off[d];
    __syncthreads();
    float s = 0.f;
    #pragma unroll
    for (int f = 0; f < 3; ++f) s += emb[(size_t)ids[f] * HID + d];
    out[(size_t)e * HID + d] = (__bf16)s;
}

// ---------------------------------------------------------------------------
// WMMA bf16 GEMM:  C[M,Nn] = act(A[M,K] @ Wt^T + bias)
//   A  : M x K  bf16 row-major
//   Wt : Nn x K bf16 row-major  (weights pre-transposed)
// 8 waves / block, each wave computes a 32x32 patch (2x2 WMMA register tile)
// -> 4 v_wmma_f32_16x16x32_bf16 per 8 b128 loads (2 loads / WMMA).
// Block tile: 128 (M) x 64 (N).
// ---------------------------------------------------------------------------
__global__ __launch_bounds__(256) void gemm_bf16(
    const __bf16* __restrict__ A, const __bf16* __restrict__ Wt,
    const float* __restrict__ bias,
    float* __restrict__ Cf, __bf16* __restrict__ Cb,
    int M, int K, int Nn, int relu)
{
    const int wave = threadIdx.x >> 5;
    const int lane = threadIdx.x & 31;
    const int half = lane >> 4;        // 0: K {0..7,16..23}, 1: K {8..15,24..31}
    const int r    = lane & 15;
    const int m0 = blockIdx.x * 128 + (wave >> 1) * 32;
    const int n0 = blockIdx.y * 64  + (wave & 1) * 32;
    if (m0 >= M) return;

    int ra0 = m0 + r;      if (ra0 > M - 1) ra0 = M - 1;   // tail-safe loads
    int ra1 = m0 + 16 + r; if (ra1 > M - 1) ra1 = M - 1;
    const __bf16* ap0 = A  + (size_t)ra0 * K + half * 8;
    const __bf16* ap1 = A  + (size_t)ra1 * K + half * 8;
    const __bf16* bp0 = Wt + (size_t)(n0 + r) * K + half * 8;
    const __bf16* bp1 = Wt + (size_t)(n0 + 16 + r) * K + half * 8;

    v8f acc00 = {0.f,0.f,0.f,0.f,0.f,0.f,0.f,0.f};
    v8f acc01 = acc00, acc10 = acc00, acc11 = acc00;

    for (int k = 0; k < K; k += 32) {
        v8bf a0l = *(const v8bf*)(ap0 + k);
        v8bf a0h = *(const v8bf*)(ap0 + k + 16);
        v8bf a1l = *(const v8bf*)(ap1 + k);
        v8bf a1h = *(const v8bf*)(ap1 + k + 16);
        v8bf b0l = *(const v8bf*)(bp0 + k);
        v8bf b0h = *(const v8bf*)(bp0 + k + 16);
        v8bf b1l = *(const v8bf*)(bp1 + k);
        v8bf b1h = *(const v8bf*)(bp1 + k + 16);
        v16bf av0 = __builtin_shufflevector(a0l, a0h, 0,1,2,3,4,5,6,7,8,9,10,11,12,13,14,15);
        v16bf av1 = __builtin_shufflevector(a1l, a1h, 0,1,2,3,4,5,6,7,8,9,10,11,12,13,14,15);
        v16bf bv0 = __builtin_shufflevector(b0l, b0h, 0,1,2,3,4,5,6,7,8,9,10,11,12,13,14,15);
        v16bf bv1 = __builtin_shufflevector(b1l, b1h, 0,1,2,3,4,5,6,7,8,9,10,11,12,13,14,15);
        acc00 = __builtin_amdgcn_wmma_f32_16x16x32_bf16(false, av0, false, bv0, (short)0, acc00, false, false);
        acc01 = __builtin_amdgcn_wmma_f32_16x16x32_bf16(false, av0, false, bv1, (short)0, acc01, false, false);
        acc10 = __builtin_amdgcn_wmma_f32_16x16x32_bf16(false, av1, false, bv0, (short)0, acc10, false, false);
        acc11 = __builtin_amdgcn_wmma_f32_16x16x32_bf16(false, av1, false, bv1, (short)0, acc11, false, false);
    }

    auto store_tile = [&](const v8f& acc, int mt, int nt) {
        const int col   = nt + r;
        const float bb  = bias ? bias[col] : 0.0f;
        const int rbase = mt + (half ? 8 : 0);
        #pragma unroll
        for (int v = 0; v < 8; ++v) {
            int row = rbase + v;
            if (row < M) {
                float xv = acc[v] + bb;
                if (relu) xv = fmaxf(xv, 0.0f);
                if (Cf) Cf[(size_t)row * Nn + col] = xv;
                if (Cb) Cb[(size_t)row * Nn + col] = (__bf16)xv;
            }
        }
    };
    store_tile(acc00, m0,      n0);
    store_tile(acc01, m0,      n0 + 16);
    store_tile(acc10, m0 + 16, n0);
    store_tile(acc11, m0 + 16, n0 + 16);
}

// ---------------------------------------------------------------------------
// Message passing: m = relu(h[src] + e_i); atomic scatter-add into aggr[dst]
// ---------------------------------------------------------------------------
__global__ __launch_bounds__(HID) void message_scatter(
    const float* __restrict__ hf, const __bf16* __restrict__ eib,
    const int* __restrict__ edge_index, float* __restrict__ aggr, int E)
{
    int e = blockIdx.x, d = threadIdx.x;
    __shared__ int sd[2];
    if (d < 2) sd[d] = edge_index[(size_t)d * E + e];
    __syncthreads();
    int src = sd[0], dst = sd[1];
    float m = hf[(size_t)src * HID + d] + (float)eib[(size_t)e * HID + d];
    m = fmaxf(m, 0.0f);
    atomicAdd(&aggr[(size_t)dst * HID + d], m);
}

// ---------------------------------------------------------------------------
// h = layer_norm(relu(z) + h)   (in-place per row, one block per node)
// ---------------------------------------------------------------------------
__global__ __launch_bounds__(HID) void ln_residual(
    float* __restrict__ hf, const float* __restrict__ zf,
    const float* __restrict__ g, const float* __restrict__ b)
{
    const int n = blockIdx.x, d = threadIdx.x;
    const size_t off = (size_t)n * HID + d;
    float y = fmaxf(zf[off], 0.0f) + hf[off];
    float s = y, s2 = y * y;
    #pragma unroll
    for (int m = 16; m >= 1; m >>= 1) {
        s  += __shfl_xor(s,  m, 32);
        s2 += __shfl_xor(s2, m, 32);
    }
    __shared__ float red[26];
    const int w = d >> 5, l = d & 31;
    if (l == 0) { red[w] = s; red[13 + w] = s2; }
    __syncthreads();
    if (d == 0) {
        float ts = 0.f, ts2 = 0.f;
        #pragma unroll
        for (int i = 0; i < 12; ++i) { ts += red[i]; ts2 += red[13 + i]; }
        red[12] = ts; red[25] = ts2;
    }
    __syncthreads();
    float mu  = red[12] * (1.0f / HID);
    float var = red[25] * (1.0f / HID) - mu * mu;
    float rs  = rsqrtf(var + 1e-5f);
    hf[off] = (y - mu) * rs * g[d] + b[d];
}

// ---------------------------------------------------------------------------
// Gate / softmax pooling / graph stats
// ---------------------------------------------------------------------------
__global__ __launch_bounds__(256) void gate_kernel(
    const float* __restrict__ hf, const float* __restrict__ gW,
    const float* __restrict__ gb, float* __restrict__ gate, int N)
{
    int n = blockIdx.x * 8 + (threadIdx.x >> 5);
    if (n >= N) return;
    int lane = threadIdx.x & 31;
    float s = 0.f;
    for (int d = lane; d < HID; d += 32) s += hf[(size_t)n * HID + d] * gW[d];
    #pragma unroll
    for (int m = 16; m >= 1; m >>= 1) s += __shfl_xor(s, m, 32);
    if (lane == 0) gate[n] = s + gb[0];
}

__device__ void atomic_max_f32(float* addr, float val) {
    int* ia = (int*)addr;
    int cur = *ia;
    while (__int_as_float(cur) < val) {
        int prev = atomicCAS(ia, cur, __float_as_int(val));
        if (prev == cur) break;
        cur = prev;
    }
}

__global__ void gmax_kernel(const float* __restrict__ gate, const int* __restrict__ batch,
                            float* __restrict__ gmax, int N) {
    int n = blockIdx.x * blockDim.x + threadIdx.x;
    if (n >= N) return;
    atomic_max_f32(&gmax[batch[n]], gate[n]);
}

__global__ void node_stats(const int* __restrict__ x, const int* __restrict__ batch,
                           const float* __restrict__ gate, const float* __restrict__ gmax,
                           float* __restrict__ gexp, float* __restrict__ gden,
                           float* __restrict__ nn, float* __restrict__ arom,
                           float* __restrict__ ring, int N) {
    int n = blockIdx.x * blockDim.x + threadIdx.x;
    if (n >= N) return;
    int g = batch[n];
    atomicAdd(&nn[g], 1.0f);
    if (x[(size_t)n * 9 + 7] == 1) atomicAdd(&arom[g], 1.0f);
    if (x[(size_t)n * 9 + 8] == 1) atomicAdd(&ring[g], 1.0f);
    float ge = expf(gate[n] - gmax[g]);
    gexp[n] = ge;
    atomicAdd(&gden[g], ge);
}

__global__ void edge_stats(const int* __restrict__ ea, const int* __restrict__ edge_index,
                           const int* __restrict__ batch, float* __restrict__ ne,
                           float* __restrict__ c1, float* __restrict__ c2,
                           float* __restrict__ c3, float* __restrict__ c12, int E) {
    int e = blockIdx.x * blockDim.x + threadIdx.x;
    if (e >= E) return;
    int g = batch[edge_index[e]];           // row 0 = src
    atomicAdd(&ne[g], 1.0f);
    int bt = ea[(size_t)e * 3];
    if      (bt == 1)  atomicAdd(&c1[g],  1.0f);
    else if (bt == 2)  atomicAdd(&c2[g],  1.0f);
    else if (bt == 3)  atomicAdd(&c3[g],  1.0f);
    else if (bt == 12) atomicAdd(&c12[g], 1.0f);
}

__global__ __launch_bounds__(HID) void pool_kernel(
    const float* __restrict__ hf, const float* __restrict__ gexp,
    const float* __restrict__ gden, const int* __restrict__ batch,
    float* __restrict__ gpool) {
    int n = blockIdx.x, d = threadIdx.x;
    int g = batch[n];
    float alpha = gexp[n] / gden[g];
    atomicAdd(&gpool[(size_t)g * HID + d], alpha * hf[(size_t)n * HID + d]);
}

__global__ __launch_bounds__(64) void graph_mlp(
    const float* nn, const float* ne, const float* arom, const float* ring,
    const float* c1, const float* c2, const float* c3, const float* c12,
    const float* __restrict__ W1, const float* __restrict__ b1,
    const float* __restrict__ W2, const float* __restrict__ b2,
    float* __restrict__ gv) {
    int g = blockIdx.x, t = threadIdx.x;
    __shared__ float f[8];
    __shared__ float hid[64];
    if (t == 0) {
        float n_ = nn[g], e_ = ne[g];
        float nc = fmaxf(n_, 1.0f), ec = fmaxf(e_, 1.0f);
        f[0] = log1pf(n_);  f[1] = log1pf(e_);
        f[2] = arom[g] / nc; f[3] = ring[g] / nc;
        f[4] = c1[g] / ec;  f[5] = c2[g] / ec;
        f[6] = c3[g] / ec;  f[7] = c12[g] / ec;
    }
    __syncthreads();
    float h = b1[t];
    #pragma unroll
    for (int i = 0; i < 8; ++i) h += f[i] * W1[i * 64 + t];
    hid[t] = fmaxf(h, 0.0f);
    __syncthreads();
    float o = b2[t];
    for (int k = 0; k < 64; ++k) o += hid[k] * W2[k * 64 + t];
    gv[(size_t)g * 64 + t] = o;
}

__global__ void concat_bf16(const float* __restrict__ gpool, const float* __restrict__ gv,
                            __bf16* __restrict__ out, int G) {
    int idx = blockIdx.x * blockDim.x + threadIdx.x;
    if (idx >= G * 448) return;
    int g = idx / 448, c = idx % 448;
    float v = (c < HID) ? gpool[(size_t)g * HID + c] : gv[(size_t)g * 64 + (c - HID)];
    out[idx] = (__bf16)v;
}

__global__ __launch_bounds__(256) void l2norm(const float* __restrict__ in,
                                              float* __restrict__ out, int G) {
    int g = blockIdx.x * 8 + (threadIdx.x >> 5);
    if (g >= G) return;
    int lane = threadIdx.x & 31;
    const float* row = in + (size_t)g * 512;
    float s = 0.f;
    for (int d = lane; d < 512; d += 32) { float v = row[d]; s += v * v; }
    #pragma unroll
    for (int m = 16; m >= 1; m >>= 1) s += __shfl_xor(s, m, 32);
    float inv = 1.0f / fmaxf(sqrtf(s), 1e-12f);
    float* o = out + (size_t)g * 512;
    for (int d = lane; d < 512; d += 32) o[d] = row[d] * inv;
}

// ---------------------------------------------------------------------------
// Host orchestration
// ---------------------------------------------------------------------------
extern "C" void kernel_launch(void* const* d_in, const int* in_sizes, int n_in,
                              void* d_out, int out_size, void* d_ws, size_t ws_size,
                              hipStream_t stream)
{
    (void)n_in; (void)ws_size;
    const int*   x          = (const int*)d_in[0];
    const int*   edge_attr  = (const int*)d_in[1];
    const int*   edge_index = (const int*)d_in[2];
    const int*   batch_idx  = (const int*)d_in[3];
    const float* atom_emb   = (const float*)d_in[4];
    const float* bond_emb   = (const float*)d_in[5];
    const float* in_proj_W  = (const float*)d_in[6];
    const float* in_proj_b  = (const float*)d_in[7];
    const float* edge_lin_W = (const float*)d_in[8];
    const float* edge_lin_b = (const float*)d_in[9];
    const float* mlp_W1     = (const float*)d_in[10];
    const float* mlp_b1     = (const float*)d_in[11];
    const float* mlp_W2     = (const float*)d_in[12];
    const float* mlp_b2     = (const float*)d_in[13];
    const float* ln_g       = (const float*)d_in[14];
    const float* ln_b       = (const float*)d_in[15];
    const float* gate_W     = (const float*)d_in[16];
    const float* gate_b     = (const float*)d_in[17];
    const float* gmlp_W1    = (const float*)d_in[18];
    const float* gmlp_b1    = (const float*)d_in[19];
    const float* gmlp_W2    = (const float*)d_in[20];
    const float* gmlp_b2    = (const float*)d_in[21];
    const float* out_W1     = (const float*)d_in[22];
    const float* out_b1     = (const float*)d_in[23];
    const float* out_W2     = (const float*)d_in[24];
    const float* out_b2     = (const float*)d_in[25];

    const int N = in_sizes[0] / 9;
    const int E = in_sizes[1] / 3;
    const int G = out_size / 512;
    const size_t NH = (size_t)N * HID;
    const size_t EH = (size_t)E * HID;
    const int sz = HID * HID;

    // ---- workspace carve-out ----
    char* p = (char*)d_ws;
    auto alloc = [&](size_t bytes) -> void* {
        void* r = (void*)p;
        p += (bytes + 255) & ~(size_t)255;
        return r;
    };
    __bf16* wInT = (__bf16*)alloc((size_t)sz * 2);
    __bf16* wEdT = (__bf16*)alloc((size_t)LAYERS * sz * 2);
    __bf16* wM1T = (__bf16*)alloc((size_t)LAYERS * sz * 2);
    __bf16* wM2T = (__bf16*)alloc((size_t)LAYERS * sz * 2);
    __bf16* wO1T = (__bf16*)alloc((size_t)HID * 448 * 2);
    __bf16* wO2T = (__bf16*)alloc((size_t)512 * HID * 2);
    float*  hf   = (float*) alloc(NH * 4);
    float*  aggr = (float*) alloc(NH * 4);
    float*  zf   = (float*) alloc(NH * 4);
    __bf16* hb   = (__bf16*)alloc(NH * 2);
    __bf16* z1b  = (__bf16*)alloc(NH * 2);
    __bf16* eb   = (__bf16*)alloc(EH * 2);
    __bf16* eib  = (__bf16*)alloc(EH * 2);
    float*  gate = (float*) alloc((size_t)N * 4);
    float*  gexp = (float*) alloc((size_t)N * 4);
    float*  gmax = (float*) alloc((size_t)G * 4);
    float*  gacc = (float*) alloc((size_t)9 * G * 4);   // gden,nn,ne,arom,ring,c1,c2,c3,c12
    float*  gpool= (float*) alloc((size_t)G * HID * 4);
    float*  gv   = (float*) alloc((size_t)G * 64 * 4);
    __bf16* aout = (__bf16*)alloc((size_t)G * 448 * 2);
    __bf16* o1b  = (__bf16*)alloc((size_t)G * HID * 2);
    float*  o2f  = (float*) alloc((size_t)G * 512 * 4);

    float *gden = gacc, *nn = gacc + G, *ne = gacc + 2*G, *arom = gacc + 3*G,
          *ring = gacc + 4*G, *c1 = gacc + 5*G, *c2 = gacc + 6*G,
          *c3 = gacc + 7*G, *c12 = gacc + 8*G;

    auto gemm = [&](const __bf16* A, const __bf16* Wt, const float* bias,
                    float* Cf, __bf16* Cb, int M, int K, int Nn, int relu) {
        dim3 grid((M + 127) / 128, Nn / 64);
        gemm_bf16<<<grid, 256, 0, stream>>>(A, Wt, bias, Cf, Cb, M, K, Nn, relu);
    };
    auto fill = [&](float* ptr, float v, size_t n) {
        fill_f32<<<(unsigned)((n + 255) / 256), 256, 0, stream>>>(ptr, v, n);
    };
    auto cvt = [&](const float* W, __bf16* Wt, int K, int Nn) {
        cvt_transpose<<<(K * Nn + 255) / 256, 256, 0, stream>>>(W, Wt, K, Nn);
    };

    // ---- weight conversion (f32 -> bf16, transposed) ----
    cvt(in_proj_W, wInT, HID, HID);
    for (int i = 0; i < LAYERS; ++i) {
        cvt(edge_lin_W + (size_t)i * sz, wEdT + (size_t)i * sz, HID, HID);
        cvt(mlp_W1     + (size_t)i * sz, wM1T + (size_t)i * sz, HID, HID);
        cvt(mlp_W2     + (size_t)i * sz, wM2T + (size_t)i * sz, HID, HID);
    }
    cvt(out_W1, wO1T, 448, HID);
    cvt(out_W2, wO2T, HID, 512);

    // ---- embeddings + input projection ----
    embed_atoms<<<N, HID, 0, stream>>>(x, atom_emb, hb, N);
    gemm(hb, wInT, in_proj_b, hf, nullptr, N, HID, HID, 1);
    embed_bonds<<<E, HID, 0, stream>>>(edge_attr, bond_emb, eb, E);

    // ---- message-passing layers ----
    for (int i = 0; i < LAYERS; ++i) {
        gemm(eb, wEdT + (size_t)i * sz, edge_lin_b + (size_t)i * HID,
             nullptr, eib, E, HID, HID, 0);
        fill(aggr, 0.0f, NH);
        message_scatter<<<E, HID, 0, stream>>>(hf, eib, edge_index, aggr, E);
        add_to_bf16<<<(unsigned)((NH + 255) / 256), 256, 0, stream>>>(hf, aggr, hb, NH);
        gemm(hb,  wM1T + (size_t)i * sz, mlp_b1 + (size_t)i * HID, nullptr, z1b, N, HID, HID, 1);
        gemm(z1b, wM2T + (size_t)i * sz, mlp_b2 + (size_t)i * HID, zf, nullptr, N, HID, HID, 0);
        ln_residual<<<N, HID, 0, stream>>>(hf, zf, ln_g + (size_t)i * HID, ln_b + (size_t)i * HID);
    }

    // ---- softmax attention pooling + graph stats ----
    gate_kernel<<<(N + 7) / 8, 256, 0, stream>>>(hf, gate_W, gate_b, gate, N);
    fill(gmax, -3.0e38f, G);
    fill(gacc, 0.0f, (size_t)9 * G);
    fill(gpool, 0.0f, (size_t)G * HID);
    gmax_kernel<<<(N + 255) / 256, 256, 0, stream>>>(gate, batch_idx, gmax, N);
    node_stats<<<(N + 255) / 256, 256, 0, stream>>>(x, batch_idx, gate, gmax,
                                                    gexp, gden, nn, arom, ring, N);
    edge_stats<<<(E + 255) / 256, 256, 0, stream>>>(edge_attr, edge_index, batch_idx,
                                                    ne, c1, c2, c3, c12, E);
    pool_kernel<<<N, HID, 0, stream>>>(hf, gexp, gden, batch_idx, gpool);
    graph_mlp<<<G, 64, 0, stream>>>(nn, ne, arom, ring, c1, c2, c3, c12,
                                    gmlp_W1, gmlp_b1, gmlp_W2, gmlp_b2, gv);

    // ---- output head ----
    concat_bf16<<<(G * 448 + 255) / 256, 256, 0, stream>>>(gpool, gv, aout, G);
    gemm(aout, wO1T, out_b1, nullptr, o1b, G, 448, HID, 1);
    gemm(o1b,  wO2T, out_b2, o2f, nullptr, G, HID, 512, 0);
    l2norm<<<(G + 7) / 8, 256, 0, stream>>>(o2f, (float*)d_out, G);
}
